// SelectiveSSM_65120294142159
// MI455X (gfx1250) — matrix operand order repaired and checked by reading, not compiled, not run on hardware
//
#include <hip/hip_runtime.h>
#include <math.h>

#define D_MODEL 1024
#define D_STATE 16
#define D_CONV  4
#define D_INNER 2048
#define BATCH   2
#define SEQLEN  1024
#define BL      (BATCH * SEQLEN)   // 2048 GEMM rows

typedef __attribute__((ext_vector_type(16))) __bf16 v16bf;
typedef __attribute__((ext_vector_type(8)))  float  v8f;

__device__ __forceinline__ float sigmoidf_(float x) { return 1.f / (1.f + __expf(-x)); }
__device__ __forceinline__ float softplusf_(float x) { return (x > 20.f) ? x : log1pf(__expf(x)); }

// ---------------------------------------------------------------------------
// fp32 -> bf16 weight conversion (once per launch, deterministic)
// ---------------------------------------------------------------------------
__global__ void cvt_bf16_kernel(const float* __restrict__ in, __bf16* __restrict__ out, int n) {
    int i = blockIdx.x * blockDim.x + threadIdx.x;
    if (i < n) out[i] = (__bf16)in[i];
}

// ---------------------------------------------------------------------------
// Load one 16x32 bf16 A-fragment (rows m0..m0+15, K = k..k+31) from fp32,
// in contiguous k order (B uses the same):
//   lanes 0-15:  K = {k..k+15}
//   lanes 16-31: K = {k+16..k+31}
// ---------------------------------------------------------------------------
__device__ __forceinline__ v16bf load_a_frag(const float* __restrict__ Arow, int k, int half) {
    const float* ap0 = Arow + k + half * 16;
    const float* ap1 = Arow + k + 8 + half * 16;
    float4 f0 = *(const float4*)(ap0);
    float4 f1 = *(const float4*)(ap0 + 4);
    float4 f2 = *(const float4*)(ap1);
    float4 f3 = *(const float4*)(ap1 + 4);
    v16bf a;
    a[0]  = (__bf16)f0.x; a[1]  = (__bf16)f0.y; a[2]  = (__bf16)f0.z; a[3]  = (__bf16)f0.w;
    a[4]  = (__bf16)f1.x; a[5]  = (__bf16)f1.y; a[6]  = (__bf16)f1.z; a[7]  = (__bf16)f1.w;
    a[8]  = (__bf16)f2.x; a[9]  = (__bf16)f2.y; a[10] = (__bf16)f2.z; a[11] = (__bf16)f2.w;
    a[12] = (__bf16)f3.x; a[13] = (__bf16)f3.y; a[14] = (__bf16)f3.z; a[15] = (__bf16)f3.w;
    return a;
}

// ---------------------------------------------------------------------------
// WMMA GEMM: C[M,N] = A[M,K](f32) * W[N,K]^T(bf16), fp32 accumulate.
// NT  = number of 16-wide N tiles per wave (compile-time)
// EPI = 1: C = softplus(C + bias[col])   (dt projection epilogue)
// Wave -> 32 x (16*NT) strip (MT=2 register blocking along M).
// Block = 256 threads = 8 waves stacked along M -> 256 x (16*NT) block tile.
// K-chunks (32 wide) are double-buffered: loads for chunk c+1 issue before the
// WMMAs of chunk c, hiding L2 latency and killing the WMMA->VALU WAR nops.
// Requires K % 64 == 0 (holds: K = 1024 or 2048).
// ---------------------------------------------------------------------------
template <int NT, int EPI>
__global__ void __launch_bounds__(256)
wmma_gemm_kernel(const float* __restrict__ A, const __bf16* __restrict__ W,
                 float* __restrict__ C, const float* __restrict__ bias,
                 int M, int N, int K) {
    const int lane  = threadIdx.x & 31;
    const int wave  = threadIdx.x >> 5;
    const int half  = lane >> 4;      // 0 or 1
    const int lidx  = lane & 15;
    const int m0    = (blockIdx.x * 8 + wave) * 32;       // 2 M-tiles per wave
    const int nbase = blockIdx.y * (16 * NT);

    const float* Arow0 = A + (size_t)(m0 + lidx) * K;
    const float* Arow1 = Arow0 + (size_t)16 * K;

    const __bf16* Wrow[NT];
#pragma unroll
    for (int i = 0; i < NT; ++i)
        Wrow[i] = W + (size_t)(nbase + i * 16 + lidx) * K + half * 16;

    v8f acc0[NT];
    v8f acc1[NT];
#pragma unroll
    for (int i = 0; i < NT; ++i) { acc0[i] = {}; acc1[i] = {}; }

    // ping-pong fragment buffers
    v16bf p_a0, p_a1, p_b[NT];
    v16bf q_a0, q_a1, q_b[NT];

    auto load_chunk = [&](int k, v16bf& a0, v16bf& a1, v16bf b[NT]) {
        a0 = load_a_frag(Arow0, k, half);
        a1 = load_a_frag(Arow1, k, half);
#pragma unroll
        for (int i = 0; i < NT; ++i) b[i] = *(const v16bf*)(Wrow[i] + k);
    };
    auto mma_chunk = [&](const v16bf& a0, const v16bf& a1, const v16bf b[NT]) {
#pragma unroll
        for (int i = 0; i < NT; ++i) {
            acc0[i] = __builtin_amdgcn_wmma_f32_16x16x32_bf16(
                false, a0, false, b[i], (short)0, acc0[i], false, false);
            acc1[i] = __builtin_amdgcn_wmma_f32_16x16x32_bf16(
                false, a1, false, b[i], (short)0, acc1[i], false, false);
        }
    };

    load_chunk(0, p_a0, p_a1, p_b);
    int k = 0;
    for (; k + 64 < K; k += 64) {
        load_chunk(k + 32, q_a0, q_a1, q_b);
        mma_chunk(p_a0, p_a1, p_b);
        load_chunk(k + 64, p_a0, p_a1, p_b);
        mma_chunk(q_a0, q_a1, q_b);
    }
    // k == K - 64: finish last two chunks, branch-free
    load_chunk(K - 32, q_a0, q_a1, q_b);
    mma_chunk(p_a0, p_a1, p_b);
    mma_chunk(q_a0, q_a1, q_b);

#pragma unroll
    for (int i = 0; i < NT; ++i) {
        int col = nbase + i * 16 + lidx;
        float bv = (EPI == 1) ? bias[col] : 0.f;
#pragma unroll
        for (int v = 0; v < 8; ++v) {
            int   row0 = m0 + half * 8 + v;
            float val0 = acc0[i][v];
            float val1 = acc1[i][v];
            if (EPI == 1) { val0 = softplusf_(val0 + bv); val1 = softplusf_(val1 + bv); }
            C[(size_t)row0 * N + col]        = val0;
            C[(size_t)(row0 + 16) * N + col] = val1;
        }
    }
}

// ---------------------------------------------------------------------------
// Causal depthwise conv1d (k=4, left pad 3) + bias + SiLU, reading xi = xz[:, :, :d_inner]
// ---------------------------------------------------------------------------
__global__ void conv_silu_kernel(const float* __restrict__ xz, const float* __restrict__ conv_w,
                                 const float* __restrict__ conv_b, float* __restrict__ u) {
    int idx = blockIdx.x * blockDim.x + threadIdx.x;
    if (idx >= BATCH * SEQLEN * D_INNER) return;
    int d = idx % D_INNER;
    int t = (idx / D_INNER) % SEQLEN;
    int b = idx / (D_INNER * SEQLEN);

    float acc = conv_b[d];
    const float* w = conv_w + d * D_CONV;
    size_t base = (size_t)b * SEQLEN * (2 * D_INNER);
#pragma unroll
    for (int j = 0; j < D_CONV; ++j) {
        int tt = t - (D_CONV - 1) + j;
        if (tt >= 0) acc += w[j] * xz[base + (size_t)tt * (2 * D_INNER) + d];
    }
    u[idx] = acc * sigmoidf_(acc);
}

// ---------------------------------------------------------------------------
// Selective scan: one thread owns one (batch, channel); 16 states in registers.
// B/C (shared by all channels) staged through LDS in 64-timestep chunks.
// ---------------------------------------------------------------------------
__global__ void scan_kernel(const float* __restrict__ dt, const float* __restrict__ u,
                            const float* __restrict__ bc, const float* __restrict__ A_log,
                            float* __restrict__ ycore) {
    __shared__ float Bs[64][16];
    __shared__ float Cs[64][16];
    const int b = blockIdx.x >> 3;                       // D_INNER/256 = 8 chunks per batch
    const int d = ((blockIdx.x & 7) << 8) + threadIdx.x;

    float Aa[D_STATE];
#pragma unroll
    for (int n = 0; n < D_STATE; ++n) Aa[n] = -__expf(A_log[(size_t)d * D_STATE + n]);
    float s[D_STATE];
#pragma unroll
    for (int n = 0; n < D_STATE; ++n) s[n] = 0.f;

    for (int t0 = 0; t0 < SEQLEN; t0 += 64) {
        __syncthreads();
#pragma unroll
        for (int j = 0; j < 8; ++j) {
            int e  = threadIdx.x * 8 + j;                // 0..2047 = 64 steps * 32 (B|C)
            int tt = e >> 5;
            int n  = e & 31;
            float v = bc[(size_t)(b * SEQLEN + t0 + tt) * (2 * D_STATE) + n];
            if (n < D_STATE) Bs[tt][n] = v;
            else             Cs[tt][n - D_STATE] = v;
        }
        __syncthreads();

        for (int tt = 0; tt < 64; ++tt) {
            size_t gi  = (size_t)(b * SEQLEN + t0 + tt) * D_INNER + d;
            float dtv = dt[gi];                          // already softplus'd
            float uv  = u[gi];
            float du  = dtv * uv;
            float y   = 0.f;
#pragma unroll
            for (int n = 0; n < D_STATE; ++n) {
                s[n] = __expf(dtv * Aa[n]) * s[n] + du * Bs[tt][n];
                y   += s[n] * Cs[tt][n];
            }
            ycore[gi] = y;
        }
    }
}

// ---------------------------------------------------------------------------
// yg = (ycore + D*u) * silu(z), z = xz[:, :, d_inner:]
// ---------------------------------------------------------------------------
__global__ void gate_kernel(const float* __restrict__ ycore, const float* __restrict__ u,
                            const float* __restrict__ Dvec, const float* __restrict__ xz,
                            float* __restrict__ yg) {
    int idx = blockIdx.x * blockDim.x + threadIdx.x;
    if (idx >= BATCH * SEQLEN * D_INNER) return;
    int d  = idx % D_INNER;
    int bl = idx / D_INNER;
    float z  = xz[(size_t)bl * (2 * D_INNER) + D_INNER + d];
    float yv = ycore[idx] + Dvec[d] * u[idx];
    yg[idx] = yv * z * sigmoidf_(z);
}

// ---------------------------------------------------------------------------
extern "C" void kernel_launch(void* const* d_in, const int* in_sizes, int n_in,
                              void* d_out, int out_size, void* d_ws, size_t ws_size,
                              hipStream_t stream) {
    (void)in_sizes; (void)n_in; (void)out_size; (void)ws_size;
    const float* x      = (const float*)d_in[0];
    const float* W_in   = (const float*)d_in[1];
    const float* conv_w = (const float*)d_in[2];
    const float* conv_b = (const float*)d_in[3];
    const float* W_x    = (const float*)d_in[4];
    const float* W_dt   = (const float*)d_in[5];
    const float* b_dt   = (const float*)d_in[6];
    const float* A_log  = (const float*)d_in[7];
    const float* Dvec   = (const float*)d_in[8];
    const float* W_out  = (const float*)d_in[9];
    float* out = (float*)d_out;

    char*  ws  = (char*)d_ws;
    size_t off = 0;
    auto take = [&](size_t bytes) -> char* {
        char* p = ws + off;
        off += (bytes + 255) & ~(size_t)255;
        return p;
    };
    __bf16* w_in_bf  = (__bf16*)take((size_t)2 * D_INNER * D_MODEL * 2);   // 8 MB
    __bf16* w_dt_bf  = (__bf16*)take((size_t)D_INNER * D_INNER * 2);       // 8 MB
    __bf16* w_x_bf   = (__bf16*)take((size_t)2 * D_STATE * D_INNER * 2);   // 128 KB
    __bf16* w_out_bf = (__bf16*)take((size_t)D_MODEL * D_INNER * 2);       // 4 MB
    float*  xz       = (float*)take((size_t)BL * 2 * D_INNER * 4);         // 32 MB
    float*  u        = (float*)take((size_t)BL * D_INNER * 4);             // 16 MB
    float*  bc       = (float*)take((size_t)BL * 2 * D_STATE * 4);         // 256 KB
    float*  dt       = (float*)take((size_t)BL * D_INNER * 4);             // 16 MB
    float*  ycore    = (float*)take((size_t)BL * D_INNER * 4);             // 16 MB
    float*  yg       = (float*)take((size_t)BL * D_INNER * 4);             // 16 MB

    // 0) weights -> bf16
    int n1 = 2 * D_INNER * D_MODEL;
    cvt_bf16_kernel<<<(n1 + 255) / 256, 256, 0, stream>>>(W_in, w_in_bf, n1);
    int n2 = D_INNER * D_INNER;
    cvt_bf16_kernel<<<(n2 + 255) / 256, 256, 0, stream>>>(W_dt, w_dt_bf, n2);
    int n3 = 2 * D_STATE * D_INNER;
    cvt_bf16_kernel<<<(n3 + 255) / 256, 256, 0, stream>>>(W_x, w_x_bf, n3);
    int n4 = D_MODEL * D_INNER;
    cvt_bf16_kernel<<<(n4 + 255) / 256, 256, 0, stream>>>(W_out, w_out_bf, n4);

    // 1) xz = x @ W_in^T   (M=2048, N=4096, K=1024)
    {
        dim3 g(BL / 256, (2 * D_INNER) / 64);
        wmma_gemm_kernel<4, 0><<<g, 256, 0, stream>>>(x, w_in_bf, xz, nullptr, BL, 2 * D_INNER, D_MODEL);
    }
    // 2) causal depthwise conv + SiLU -> u
    int ne = BATCH * SEQLEN * D_INNER;
    conv_silu_kernel<<<(ne + 255) / 256, 256, 0, stream>>>(xz, conv_w, conv_b, u);
    // 3) BC = u @ W_x^T    (M=2048, N=32, K=2048)  -> NT=2 covers all 32 cols
    {
        dim3 g(BL / 256, 1);
        wmma_gemm_kernel<2, 0><<<g, 256, 0, stream>>>(u, w_x_bf, bc, nullptr, BL, 2 * D_STATE, D_INNER);
    }
    // 4) dt = softplus(u @ W_dt^T + b_dt)   (M=2048, N=2048, K=2048)
    {
        dim3 g(BL / 256, D_INNER / 64);
        wmma_gemm_kernel<4, 1><<<g, 256, 0, stream>>>(u, w_dt_bf, dt, b_dt, BL, D_INNER, D_INNER);
    }
    // 5) selective scan -> ycore
    scan_kernel<<<BATCH * (D_INNER / 256), 256, 0, stream>>>(dt, u, bc, A_log, ycore);
    // 6) gate: yg = (ycore + D*u) * silu(z)
    gate_kernel<<<(ne + 255) / 256, 256, 0, stream>>>(ycore, u, Dvec, xz, yg);
    // 7) out = yg @ W_out^T   (M=2048, N=1024, K=2048)
    {
        dim3 g(BL / 256, D_MODEL / 64);
        wmma_gemm_kernel<4, 0><<<g, 256, 0, stream>>>(yg, w_out_bf, out, nullptr, BL, D_MODEL, D_INNER);
    }
}
